// CrossAttentionLayer_48232482734762
// MI455X (gfx1250) — matrix operand order repaired and checked by reading, not compiled, not run on hardware
//
#include <hip/hip_runtime.h>

// Problem sizes (fixed by the reference)
constexpr int Bn = 2;     // batch
constexpr int Lq = 2048;  // query len
constexpr int Lk = 2048;  // key len
constexpr int Hh = 1024;  // hidden
constexpr int Nh = 16;    // heads
constexpr int Dh = 64;    // head dim

typedef __bf16 bf16;
typedef __attribute__((ext_vector_type(16))) __bf16 v16bf;
typedef __attribute__((ext_vector_type(8)))  float  v8f;

typedef unsigned int uint32x4 __attribute__((ext_vector_type(4)));
typedef int          int32x4  __attribute__((ext_vector_type(4)));
typedef int          int32x8  __attribute__((ext_vector_type(8)));

// async-to-LDS builtin takes int4-vector pointers in explicit address spaces
typedef __attribute__((address_space(1))) int32x4 g_i32x4;  // global
typedef __attribute__((address_space(3))) int32x4 l_i32x4;  // LDS
typedef __attribute__((address_space(3))) void    las_void; // LDS (addr calc)

// CDNA5 feature probes (false on the host pass -> fallbacks parse there)
#if defined(__has_builtin)
#  if __has_builtin(__builtin_amdgcn_global_load_async_to_lds_b128)
#    define HAVE_ASYNC_LDS 1
#  endif
#  if __has_builtin(__builtin_amdgcn_tensor_load_to_lds) && \
      __has_builtin(__builtin_amdgcn_s_wait_tensorcnt)
#    define HAVE_TDM 1
#  endif
#endif

__device__ __forceinline__ void wait_async0() {
#if defined(__has_builtin) && __has_builtin(__builtin_amdgcn_s_wait_asynccnt)
  __builtin_amdgcn_s_wait_asynccnt(0);
#else
  asm volatile("s_wait_asynccnt 0" ::: "memory");
#endif
}

// ---------------------------------------------------------------------------
// WMMA fragment loaders (CDNA5 16x16x32 bf16 layouts, ISA 7.12.2)
// A (16Mx32K): lane l -> row m=l&15; elems 0..7 = K[(l>>4)*8 .. +7],
//              elems 8..15 = K[16+(l>>4)*8 .. +7]  (two 16B contiguous chunks)
// B (32Kx16N): lane l -> col n=l&15; elems 0..7 = K[(l>>4)*8 .. +7],
//              elems 8..15 = K[16+(l>>4)*8 .. +7]  (strided by ldb)
// C/D (16x16): elem r -> m = r + 8*(l>>4); n = l&15
// ---------------------------------------------------------------------------
__device__ __forceinline__ v16bf load_a_bf16(const bf16* __restrict__ base, int lda, int k0) {
  const int lane = threadIdx.x & 31;
  const int hf = lane >> 4, m = lane & 15;
  const bf16* p = base + (size_t)m * lda + k0 + hf * 8;
  v16bf a;
#pragma unroll
  for (int i = 0; i < 8; ++i) { a[i] = p[i]; a[i + 8] = p[i + 16]; }
  return a;
}

// A-fragment from bf16 LDS strip (pointer keeps LDS addrspace -> ds_load)
__device__ __forceinline__ v16bf load_a_lds_bf16(const bf16* base, int lda, int k0) {
  const int lane = threadIdx.x & 31;
  const int hf = lane >> 4, m = lane & 15;
  const bf16* p = base + m * lda + k0 + hf * 8;
  v16bf a;
#pragma unroll
  for (int i = 0; i < 8; ++i) { a[i] = p[i]; a[i + 8] = p[i + 16]; }
  return a;
}

// A-fragment from fp32 LDS strip, converting to bf16
__device__ __forceinline__ v16bf load_a_lds_f32(const float* base, int lda, int k0) {
  const int lane = threadIdx.x & 31;
  const int hf = lane >> 4, m = lane & 15;
  const float* p = base + m * lda + k0 + hf * 8;
  v16bf a;
#pragma unroll
  for (int i = 0; i < 8; ++i) { a[i] = (bf16)p[i]; a[i + 8] = (bf16)p[i + 16]; }
  return a;
}

__device__ __forceinline__ v16bf load_b_nc(const bf16* __restrict__ base, int ldb, int k0, int n0) {
  const int lane = threadIdx.x & 31;
  const int hf = lane >> 4;
  const bf16* p = base + (size_t)(k0 + hf * 8) * ldb + n0 + (lane & 15);
  v16bf b;
#pragma unroll
  for (int e = 0; e < 8; ++e) { b[e] = p[(size_t)e * ldb]; b[e + 8] = p[(size_t)(e + 16) * ldb]; }
  return b;
}

__device__ __forceinline__ v8f wmma_bf16(v16bf a, v16bf b, v8f c) {
  return __builtin_amdgcn_wmma_f32_16x16x32_bf16(false, a, false, b, (short)0, c, false, false);
}

// half-wave (16-lane) reductions, wave32-native
__device__ __forceinline__ float hmax16(float v) {
#pragma unroll
  for (int o = 8; o > 0; o >>= 1) v = fmaxf(v, __shfl_xor(v, o, 16));
  return v;
}
__device__ __forceinline__ float hsum16(float v) {
#pragma unroll
  for (int o = 8; o > 0; o >>= 1) v += __shfl_xor(v, o, 16);
  return v;
}

// ---------------------------------------------------------------------------
// Stage a contiguous 16 x Hh bf16 A-strip (32 KB) into LDS.
// Preferred path: one TDM descriptor issued by wave 0 (tensor_load_to_lds),
// tracked by TENSORcnt. Fallback: cooperative 16B copies.
// ---------------------------------------------------------------------------
__device__ __forceinline__ void stage_a_strip(const bf16* __restrict__ gsrc, bf16* lds) {
#if defined(HAVE_TDM)
  if (threadIdx.x < 32) {
    const unsigned lds_addr = (unsigned)(size_t)(las_void*)lds;
    const unsigned long long ga = (unsigned long long)(size_t)gsrc;
    // D# group0: count=1 | lds_addr | global_addr | type=2 ("image")
    uint32x4 g0;
    g0[0] = 1u;
    g0[1] = lds_addr;
    g0[2] = (unsigned)ga;
    g0[3] = ((unsigned)(ga >> 32) & 0x01FFFFFFu) | (2u << 30);
    // D# group1: data_size=2B; tensor_dim0 = 1<<20 (no OOB), tensor_dim1=1;
    //            tile_dim0 = 16*Hh (1-D copy), tile_dim1/2 = 0 (unused)
    int32x8 g1;
    g1[0] = (int)(1u << 16);        // data_size = 1 -> 2 bytes
    g1[1] = 0;                      // tensor_dim0[15:0] (=0 of 1<<20)
    g1[2] = (int)0x00010010u;       // tensor_dim0[31:16]=0x10, tensor_dim1=1
    g1[3] = (int)((unsigned)(16 * Hh) << 16);  // tile_dim0 = 16384
    g1[4] = 0;                      // tile_dim1=0, tile_dim2=0
    g1[5] = (int)(1u << 20);        // tensor_dim0_stride[31:0]
    g1[6] = 0;
    g1[7] = 0;
    int32x4 gz = {0, 0, 0, 0};
#if __clang_major__ >= 23
    int32x8 gz8 = {0, 0, 0, 0, 0, 0, 0, 0};
    __builtin_amdgcn_tensor_load_to_lds(g0, g1, gz, gz, gz8, 0);
#else
    __builtin_amdgcn_tensor_load_to_lds(g0, g1, gz, gz, 0);
#endif
    __builtin_amdgcn_s_wait_tensorcnt(0);
  }
  __syncthreads();
#else
  const uint4* s = (const uint4*)gsrc;
  uint4* d = (uint4*)lds;
  for (int i = threadIdx.x; i < (16 * Hh) / 8; i += 256) d[i] = s[i];
  __syncthreads();
#endif
}

// ---------------------------------------------------------------------------
// 1) fp32 -> bf16 elementwise convert
// ---------------------------------------------------------------------------
__global__ void ca_cvt_bf16(const float* __restrict__ src, bf16* __restrict__ dst, int n) {
  int i = blockIdx.x * blockDim.x + threadIdx.x;
  const int st = gridDim.x * blockDim.x;
  for (; i < n; i += st) dst[i] = (bf16)src[i];
}

// ---------------------------------------------------------------------------
// 2) Weight transpose-convert: dst[h][o] = (bf16) src[o][h].
//    fp32 tile staged in LDS with async copies (ASYNCcnt) when available.
//    Row stride 36 floats = 144 B keeps 16B alignment for B128 LDS writes.
// ---------------------------------------------------------------------------
__global__ void ca_wtrans_bf16(const float* __restrict__ src, bf16* __restrict__ dst) {
  __shared__ __align__(16) float ftile[32][36];
  const int o0 = blockIdx.x * 32, h0 = blockIdx.y * 32;
  const int tx = threadIdx.x, ty = threadIdx.y;
  const int t = ty * 32 + tx;                // 0..255
  const int r = t >> 3, c4 = (t & 7) * 4;    // each thread: one 16B chunk
#if defined(HAVE_ASYNC_LDS)
  __builtin_amdgcn_global_load_async_to_lds_b128(
      (g_i32x4*)(src + (size_t)(o0 + r) * Hh + h0 + c4),
      (l_i32x4*)&ftile[r][c4], 0, 0);
  wait_async0();
#else
  *(float4*)&ftile[r][c4] = *(const float4*)(src + (size_t)(o0 + r) * Hh + h0 + c4);
#endif
  __syncthreads();
#pragma unroll
  for (int rr = ty; rr < 32; rr += 8)
    dst[(size_t)(h0 + rr) * Hh + (o0 + tx)] = (bf16)ftile[tx][rr];
}

// ---------------------------------------------------------------------------
// 3) QKV projection GEMM (bf16 WMMA, fp32 acc), A-strip staged via TDM:
//    MODE 0: Q -> [b,h,q,d] * (1/sqrt(Dh))
//    MODE 1: K -> [b,h,d,k]   MODE 2: V -> [b,h,k,d]
// ---------------------------------------------------------------------------
template <int MODE>
__global__ void ca_gemm_qkv(const bf16* __restrict__ X, const bf16* __restrict__ WT,
                            bf16* __restrict__ out, float scale) {
  __shared__ __align__(16) bf16 Alds[16 * Hh];   // 32 KB
  const int wave = threadIdx.x >> 5;
  const int n0 = blockIdx.x * 512 + wave * 64;
  const int m0 = blockIdx.y * 16;

  stage_a_strip(X + (size_t)m0 * Hh, Alds);

  v8f acc[4] = {};
  for (int k0 = 0; k0 < Hh; k0 += 32) {
    v16bf a = load_a_lds_bf16(Alds, Hh, k0);
#pragma unroll
    for (int j = 0; j < 4; ++j) {
      v16bf b = load_b_nc(WT, Hh, k0, n0 + j * 16);
      acc[j] = wmma_bf16(a, b, acc[j]);
    }
  }

  const int lane = threadIdx.x & 31;
  const int nl = lane & 15, hf = lane >> 4;
#pragma unroll
  for (int j = 0; j < 4; ++j) {
    const int gn = n0 + j * 16 + nl;
    const int h = gn >> 6, d = gn & 63;
#pragma unroll
    for (int r = 0; r < 8; ++r) {
      const int gm = m0 + r + 8 * hf;
      const int bb = gm >> 11;
      const int s  = gm & 2047;
      const float v = acc[j][r] * scale;
      size_t idx;
      if (MODE == 0)      idx = (((size_t)(bb * Nh + h) * Lq) + s) * Dh + d;
      else if (MODE == 1) idx = (((size_t)(bb * Nh + h) * Dh) + d) * Lk + s;
      else                idx = (((size_t)(bb * Nh + h) * Lk) + s) * Dh + d;
      out[idx] = (bf16)v;
    }
  }
}

// ---------------------------------------------------------------------------
// 4) FUSED attention: scores -> softmax -> weights(NT) -> P@V, one kernel.
//    One block (8 wave32) per (bh, 16-query tile).
//    LDS: S strip 16 x Lk fp32 (128 KB) + 8x16x64 partial ctx (32 KB).
// ---------------------------------------------------------------------------
__global__ void ca_attn_fused(const bf16* __restrict__ Q, const bf16* __restrict__ Kt,
                              const bf16* __restrict__ V, const int* __restrict__ mask,
                              float* __restrict__ Wgt, bf16* __restrict__ Ctx) {
  extern __shared__ char smem_raw[];
  float* S    = (float*)smem_raw;            // [16][Lk]
  float* part = S + 16 * Lk;                 // [8][16][Dh]

  const int bh = blockIdx.y;
  const int bb = bh / Nh, h = bh % Nh;
  const int q0 = blockIdx.x * 16;
  const int t = threadIdx.x;
  const int wave = t >> 5;
  const int lane = t & 31;
  const int nl = lane & 15, hf = lane >> 4;

  const bf16* Abase = Q  + ((size_t)bh * Lq + q0) * Dh;
  const bf16* Kb    = Kt + (size_t)bh * Dh * Lk;
  const bf16* Vb    = V  + (size_t)bh * Lk * Dh;

  // ---- Phase 1: S[16][Lk] = Q . K^T (masked) into LDS -------------------
  const v16bf a0 = load_a_bf16(Abase, Dh, 0);
  const v16bf a1 = load_a_bf16(Abase, Dh, 32);

#pragma unroll
  for (int c = 0; c < 4; ++c) {              // 4 chunks of 64 keys per wave
    const int n0 = wave * 256 + c * 64;
    v8f acc[4] = {};
#pragma unroll
    for (int j = 0; j < 4; ++j) {
      v16bf b = load_b_nc(Kb, Lk, 0, n0 + j * 16);
      acc[j] = wmma_bf16(a0, b, acc[j]);
      b = load_b_nc(Kb, Lk, 32, n0 + j * 16);
      acc[j] = wmma_bf16(a1, b, acc[j]);
    }
#pragma unroll
    for (int j = 0; j < 4; ++j) {
      const int kp = n0 + j * 16 + nl;
      const bool keep = (mask[bb * Lk + kp] != 0);
#pragma unroll
      for (int r = 0; r < 8; ++r)
        S[(r + 8 * hf) * Lk + kp] = keep ? acc[j][r] : -__builtin_inff();
    }
  }
  __syncthreads();

  // ---- Phase 2: row softmax in LDS (16 threads per row, half-wave shfl) --
  {
    const int row = t >> 4;                  // 0..15
    const int tt  = t & 15;
    float* Sr = S + row * Lk;

    float mx = -__builtin_inff();
#pragma unroll 8
    for (int i = 0; i < Lk / 16; ++i) mx = fmaxf(mx, Sr[tt + i * 16]);
    mx = hmax16(mx);

    float sum = 0.f;
#pragma unroll 8
    for (int i = 0; i < Lk / 16; ++i) {
      const float e = __expf(Sr[tt + i * 16] - mx);
      Sr[tt + i * 16] = e;
      sum += e;
    }
    sum = hsum16(sum);
    const float inv = 1.0f / sum;
#pragma unroll 8
    for (int i = 0; i < Lk / 16; ++i) Sr[tt + i * 16] *= inv;
  }
  __syncthreads();

  // ---- Phase 2b: coalesced non-temporal dump of weights (write-once) -----
  {
    float* Wbase = Wgt + ((size_t)bh * Lq + q0) * Lk;   // 16*Lk contiguous
#pragma unroll 4
    for (int idx = t; idx < 16 * Lk; idx += 256)
      __builtin_nontemporal_store(S[idx], Wbase + idx);
  }

  // ---- Phase 3: partial P@V per wave over its 256-key slice --------------
  {
    v8f acc[4] = {};
    for (int k0 = wave * 256; k0 < wave * 256 + 256; k0 += 32) {
      v16bf a = load_a_lds_f32(S, Lk, k0);   // ds_load + cvt to bf16
#pragma unroll
      for (int j = 0; j < 4; ++j) {
        v16bf b = load_b_nc(Vb, Dh, k0, j * 16);
        acc[j] = wmma_bf16(a, b, acc[j]);
      }
    }
#pragma unroll
    for (int j = 0; j < 4; ++j)
#pragma unroll
      for (int r = 0; r < 8; ++r)
        part[(wave * 16 + r + 8 * hf) * Dh + j * 16 + nl] = acc[j][r];
  }
  __syncthreads();

  // ---- Phase 3b: reduce 8 wave partials, store bf16 context --------------
#pragma unroll
  for (int i = 0; i < 4; ++i) {
    const int o = t + i * 256;               // 0..1023 over 16x64
    const int row = o >> 6, d = o & 63;
    float s = 0.f;
#pragma unroll
    for (int w = 0; w < 8; ++w) s += part[(w * 16 + row) * Dh + d];
    Ctx[((size_t)bb * Lq + q0 + row) * Hh + h * 64 + d] = (bf16)s;
  }
}

// ---------------------------------------------------------------------------
// 5) Output projection: Out = Ctx @ Wo^T + bo (A-strip staged via TDM)
// ---------------------------------------------------------------------------
__global__ void ca_outproj(const bf16* __restrict__ Ctx, const bf16* __restrict__ WoT,
                           const float* __restrict__ bo, float* __restrict__ Out) {
  __shared__ __align__(16) bf16 Alds[16 * Hh];   // 32 KB
  const int wave = threadIdx.x >> 5;
  const int n0 = blockIdx.x * 512 + wave * 64;
  const int m0 = blockIdx.y * 16;

  stage_a_strip(Ctx + (size_t)m0 * Hh, Alds);

  v8f acc[4] = {};
  for (int k0 = 0; k0 < Hh; k0 += 32) {
    v16bf a = load_a_lds_bf16(Alds, Hh, k0);
#pragma unroll
    for (int j = 0; j < 4; ++j) {
      v16bf b = load_b_nc(WoT, Hh, k0, n0 + j * 16);
      acc[j] = wmma_bf16(a, b, acc[j]);
    }
  }

  const int lane = threadIdx.x & 31;
  const int nl = lane & 15, hf = lane >> 4;
#pragma unroll
  for (int j = 0; j < 4; ++j) {
    const int gn = n0 + j * 16 + nl;
    const float bias = bo[gn];
#pragma unroll
    for (int r = 0; r < 8; ++r)
      Out[(size_t)(m0 + r + 8 * hf) * Hh + gn] = acc[j][r] + bias;
  }
}

// ---------------------------------------------------------------------------
extern "C" void kernel_launch(void* const* d_in, const int* in_sizes, int n_in,
                              void* d_out, int out_size, void* d_ws, size_t ws_size,
                              hipStream_t stream) {
  const float* dec  = (const float*)d_in[0];  // [B,Lq,H]
  const float* enc  = (const float*)d_in[1];  // [B,Lk,H]
  const int*   mask = (const int*)d_in[2];    // [B,Lk]
  const float* Wq   = (const float*)d_in[3];  // [H,H]
  const float* Wk   = (const float*)d_in[4];
  const float* Wv   = (const float*)d_in[5];
  const float* Wo   = (const float*)d_in[6];
  const float* bo   = (const float*)d_in[7];  // [H]

  float* Out = (float*)d_out;                 // [B,Lq,H]
  float* Wgt = Out + (size_t)Bn * Lq * Hh;    // [B,Nh,Lq,Lk]

  const size_t nX = (size_t)Bn * Lq * Hh;     // 4M elems
  const size_t nW = (size_t)Hh * Hh;          // 1M elems
  char* w = (char*)d_ws;
  bf16* XdB  = (bf16*)w; w += nX * 2;
  bf16* XeB  = (bf16*)w; w += nX * 2;
  bf16* WqT  = (bf16*)w; w += nW * 2;
  bf16* WkT  = (bf16*)w; w += nW * 2;
  bf16* WvT  = (bf16*)w; w += nW * 2;
  bf16* WoT  = (bf16*)w; w += nW * 2;
  bf16* Qb   = (bf16*)w; w += nX * 2;   // [B,Nh,Lq,Dh] (pre-scaled)
  bf16* Ktb  = (bf16*)w; w += nX * 2;   // [B,Nh,Dh,Lk]
  bf16* Vb   = (bf16*)w; w += nX * 2;   // [B,Nh,Lk,Dh]
  bf16* CtxB = (bf16*)w; w += nX * 2;   // [B,Lq,H]

  const dim3 blk(256);

  ca_cvt_bf16<<<2048, blk, 0, stream>>>(dec, XdB, (int)nX);
  ca_cvt_bf16<<<2048, blk, 0, stream>>>(enc, XeB, (int)nX);

  const dim3 tb(32, 8), tg(Hh / 32, Hh / 32);
  ca_wtrans_bf16<<<tg, tb, 0, stream>>>(Wq, WqT);
  ca_wtrans_bf16<<<tg, tb, 0, stream>>>(Wk, WkT);
  ca_wtrans_bf16<<<tg, tb, 0, stream>>>(Wv, WvT);
  ca_wtrans_bf16<<<tg, tb, 0, stream>>>(Wo, WoT);

  ca_gemm_qkv<0><<<dim3(Hh / 512, (Bn * Lq) / 16), blk, 0, stream>>>(XdB, WqT, Qb, 0.125f);
  ca_gemm_qkv<1><<<dim3(Hh / 512, (Bn * Lk) / 16), blk, 0, stream>>>(XeB, WkT, Ktb, 1.0f);
  ca_gemm_qkv<2><<<dim3(Hh / 512, (Bn * Lk) / 16), blk, 0, stream>>>(XeB, WvT, Vb, 1.0f);

  // fused attention: 128 KB score strip + 32 KB ctx partials in LDS
  const size_t smem = (size_t)(16 * Lk + 8 * 16 * Dh) * sizeof(float);
  ca_attn_fused<<<dim3(Lq / 16, Bn * Nh), blk, smem, stream>>>(Qb, Ktb, Vb, mask, Wgt, CtxB);

  ca_outproj<<<dim3(Hh / 512, (Bn * Lq) / 16), blk, 0, stream>>>(CtxB, WoT, bo, Out);
}
